// BipartiteMHXA_85383949844814
// MI455X (gfx1250) — matrix operand order repaired and checked by reading, not compiled, not run on hardware
//
#include <hip/hip_runtime.h>
#include <hip/hip_bf16.h>

#define NQ_   100000
#define NKV_  100000
#define NE_   2000000
#define DIM_  64
#define NH_   4
#define HF_   16

typedef __attribute__((ext_vector_type(16))) _Float16 v16h;
typedef __attribute__((ext_vector_type(8)))  float    v8f;

// ---------------------------------------------------------------------------
// WMMA GEMM: Y[M x NCOLS] = X[M x 64] @ W[64 x NCOLS] (+ bias)
// W (64 x NCOLS, f32) is staged once per workgroup into LDS with the CDNA5
// async global->LDS engine (ASYNCcnt), then each wave builds its B fragments
// from LDS. One wave per 16x16 output tile; waves cover the N dimension.
// K=64 handled as two 16x16x32 f16 WMMAs accumulating in f32.
// ---------------------------------------------------------------------------
template<int NCOLS, bool ADD_BIAS>
__global__ __launch_bounds__(256) void gemm16_wmma(const float* __restrict__ X,
                                                   const float* __restrict__ W,
                                                   const float* __restrict__ bias,
                                                   float* __restrict__ Y,
                                                   int Mrows) {
  __shared__ __attribute__((aligned(16))) float Ws[64 * NCOLS];

  const int wave = threadIdx.x >> 5;      // N-tile index within block
  const int lane = threadIdx.x & 31;
  const int m0   = blockIdx.x << 4;
  const int n0   = wave << 4;
  if (m0 >= Mrows) return;                // uniform per block (grid exact)

  // ---- async stage W into LDS (16B per lane per issue; ASYNCcnt-tracked) ----
  {
    const int nthr = blockDim.x;
    for (int base = threadIdx.x * 4; base < 64 * NCOLS; base += nthr * 4) {
      const unsigned lds_off = (unsigned)(size_t)(&Ws[base]);
      const float*   gsrc    = W + base;
      asm volatile("global_load_async_to_lds_b128 %0, %1, off"
                   :: "v"(lds_off), "v"(gsrc) : "memory");
    }
    asm volatile("s_wait_asynccnt 0x0" ::: "memory");
    __syncthreads();
  }

  const int hs = lane >> 4;               // half-wave select (0/1)
  const int lm = lane & 15;

  v8f c = {};
#pragma unroll
  for (int kc = 0; kc < 64; kc += 32) {
    // A tile: rows m0..m0+15, cols kc..kc+31.  Lane holds row M=lm.
    // element i -> K(within chunk) = hs*16 + i
    const float* xrow = X + (size_t)(m0 + lm) * 64 + kc;
    v16h a, b;
#pragma unroll
    for (int i = 0; i < 16; ++i) {
      const int ka = (hs << 4) + i;
      a[i] = (_Float16)xrow[ka];
    }
    // B tile from LDS: rows (K) kc..kc+31 of W, cols n0..n0+15.
    // N = lm, K = kc + hs*16 + i.
#pragma unroll
    for (int i = 0; i < 16; ++i) {
      const int kb = kc + (hs << 4) + i;
      b[i] = (_Float16)Ws[kb * NCOLS + n0 + lm];
    }
    c = __builtin_amdgcn_wmma_f32_16x16x32_f16(false, a, false, b,
                                               (short)0, c, false, false);
  }
  // C/D layout: element r -> M = r + hs*8, N = lm
  const float bv = ADD_BIAS ? bias[n0 + lm] : 0.0f;
#pragma unroll
  for (int r = 0; r < 8; ++r) {
    const int m = m0 + r + (hs << 3);
    Y[(size_t)m * NCOLS + n0 + lm] = c[r] + bv;
  }
}

// ---------------------------------------------------------------------------
// Workspace init (segment max / sum / output accumulators)
// ---------------------------------------------------------------------------
__global__ void init_buffers(float* __restrict__ smax,
                             float* __restrict__ ssum,
                             float* __restrict__ acc) {
  const int i = blockIdx.x * blockDim.x + threadIdx.x;
  if (i < NQ_ * NH_) { smax[i] = -3.0e38f; ssum[i] = 0.0f; }
  if (i < NQ_ * DIM_) acc[i] = 0.0f;
}

__device__ __forceinline__ void atomic_max_float(float* addr, float v) {
  // Non-negative floats order like ints; negative ones inversely like uints.
  if (v >= 0.0f) atomicMax((int*)addr, __float_as_int(v));
  else           atomicMin((unsigned int*)addr, __float_as_uint(v));
}

// ---------------------------------------------------------------------------
// Pass 1: per-(edge,head) score = <q[t],k[s]>/sqrt(F); segment atomic max
// ---------------------------------------------------------------------------
__global__ void edge_score_max(const float* __restrict__ q,
                               const float* __restrict__ kv,
                               const int* __restrict__ t,
                               const int* __restrict__ s,
                               float* __restrict__ score,
                               float* __restrict__ smax) {
  const long idx = (long)blockIdx.x * blockDim.x + threadIdx.x;  // e*4+h
  if (idx >= (long)NE_ * NH_) return;
  const int e = (int)(idx >> 2);
  const int h = (int)(idx & 3);
  const int ti = t[e], si = s[e];
  const float4* qp = (const float4*)(q  + (size_t)ti * DIM_ + h * HF_);
  const float4* kp = (const float4*)(kv + (size_t)si * (2 * DIM_) + h * HF_);
  float acc = 0.0f;
#pragma unroll
  for (int i = 0; i < 4; ++i) {
    const float4 qa = qp[i], ka = kp[i];
    acc += qa.x * ka.x + qa.y * ka.y + qa.z * ka.z + qa.w * ka.w;
  }
  acc *= 0.25f;  // 1/sqrt(16)
  score[idx] = acc;
  atomic_max_float(smax + (size_t)ti * NH_ + h, acc);
}

// ---------------------------------------------------------------------------
// Pass 2: ex = exp(score - smax[t]); segment atomic sum (score buffer reused)
// ---------------------------------------------------------------------------
__global__ void edge_exp_sum(const int* __restrict__ t,
                             float* __restrict__ score,
                             const float* __restrict__ smax,
                             float* __restrict__ ssum) {
  const long idx = (long)blockIdx.x * blockDim.x + threadIdx.x;
  if (idx >= (long)NE_ * NH_) return;
  const int e = (int)(idx >> 2);
  const int h = (int)(idx & 3);
  const int ti = t[e];
  const float ex = __expf(score[idx] - smax[(size_t)ti * NH_ + h]);
  score[idx] = ex;
  atomicAdd(ssum + (size_t)ti * NH_ + h, ex);
}

// ---------------------------------------------------------------------------
// Pass 3: alpha = ex/ssum[t]; acc[t,h,:] += alpha * v[s,h,:]
// ---------------------------------------------------------------------------
__global__ void edge_scatter_v(const float* __restrict__ kv,
                               const int* __restrict__ t,
                               const int* __restrict__ s,
                               const float* __restrict__ ex,
                               const float* __restrict__ ssum,
                               float* __restrict__ acc) {
  const long idx = (long)blockIdx.x * blockDim.x + threadIdx.x;
  if (idx >= (long)NE_ * NH_) return;
  const int e = (int)(idx >> 2);
  const int h = (int)(idx & 3);
  const int ti = t[e], si = s[e];
  const float alpha = ex[idx] / ssum[(size_t)ti * NH_ + h];
  const float4* vp = (const float4*)(kv + (size_t)si * (2 * DIM_) + DIM_ + h * HF_);
  float* op = acc + (size_t)ti * DIM_ + h * HF_;
#pragma unroll
  for (int i = 0; i < 4; ++i) {
    const float4 vv = vp[i];
    atomicAdd(op + i * 4 + 0, alpha * vv.x);
    atomicAdd(op + i * 4 + 1, alpha * vv.y);
    atomicAdd(op + i * 4 + 2, alpha * vv.z);
    atomicAdd(op + i * 4 + 3, alpha * vv.w);
  }
}

// ---------------------------------------------------------------------------
extern "C" void kernel_launch(void* const* d_in, const int* in_sizes, int n_in,
                              void* d_out, int out_size, void* d_ws, size_t ws_size,
                              hipStream_t stream) {
  (void)in_sizes; (void)n_in; (void)out_size; (void)ws_size;
  const float* input = (const float*)d_in[0];
  const float* other = (const float*)d_in[1];
  const int*   t     = (const int*)  d_in[2];
  const int*   s     = (const int*)  d_in[3];
  const float* Wq    = (const float*)d_in[4];
  const float* Wkv   = (const float*)d_in[5];
  const float* Wo    = (const float*)d_in[6];
  const float* bo    = (const float*)d_in[7];
  float* out = (float*)d_out;

  // Workspace layout (floats): ~34.4M floats (~138 MB) -> L2-resident on MI455X
  float* ws   = (float*)d_ws;
  float* qb   = ws;                               // NQ*64
  float* kvb  = qb   + (size_t)NQ_ * DIM_;        // NKV*128 (k | v)
  float* sc   = kvb  + (size_t)NKV_ * 2 * DIM_;   // E*4 (scores, then ex)
  float* smax = sc   + (size_t)NE_ * NH_;         // NQ*4
  float* ssum = smax + (size_t)NQ_ * NH_;         // NQ*4
  float* accb = ssum + (size_t)NQ_ * NH_;         // NQ*64

  init_buffers<<<(NQ_ * DIM_ + 255) / 256, 256, 0, stream>>>(smax, ssum, accb);

  // Projections via WMMA (f16 inputs, f32 accumulation), W staged via async LDS
  gemm16_wmma<DIM_, false><<<NQ_ / 16, 128, 0, stream>>>(input, Wq, nullptr, qb, NQ_);
  gemm16_wmma<2 * DIM_, false><<<NKV_ / 16, 256, 0, stream>>>(other, Wkv, nullptr, kvb, NKV_);

  const int eblocks = (int)(((long)NE_ * NH_ + 255) / 256);
  edge_score_max<<<eblocks, 256, 0, stream>>>(qb, kvb, t, s, sc, smax);
  edge_exp_sum  <<<eblocks, 256, 0, stream>>>(t, sc, smax, ssum);
  edge_scatter_v<<<eblocks, 256, 0, stream>>>(kvb, t, s, sc, ssum, accb);

  // Output projection + bias via WMMA
  gemm16_wmma<DIM_, true><<<NQ_ / 16, 128, 0, stream>>>(accb, Wo, bo, out, NQ_);
}
